// MultiHeadAttention_65266323030062
// MI455X (gfx1250) — matrix operand order repaired and checked by reading, not compiled, not run on hardware
//
#include <hip/hip_runtime.h>

// ---------------------------------------------------------------------------
// MHA forward for MI455X (gfx1250, wave32, WMMA).
// All GEMMs + attention matmuls run on v_wmma_f32_16x16x32_bf16.
// GEMM uses CDNA5 async global->LDS copies (ASYNCcnt) with double buffering
// when the toolchain exposes the builtin; otherwise the sync path.
// ---------------------------------------------------------------------------

typedef __attribute__((ext_vector_type(16))) __bf16 bf16x16;
typedef __attribute__((ext_vector_type(8)))  float  f32x8;

#if defined(__gfx1250__) && __has_builtin(__builtin_amdgcn_global_load_async_to_lds_b128)
#define USE_ASYNC_LDS 1
#else
#define USE_ASYNC_LDS 0
#endif

union Frag16 {
  bf16x16 v;
  uint4   q[2];
};

__device__ __forceinline__ unsigned short f2bf_u(float f) {
  unsigned u = __builtin_bit_cast(unsigned, f);
  unsigned r = u + 0x7FFFu + ((u >> 16) & 1u);   // round-to-nearest-even
  return (unsigned short)(r >> 16);
}
__device__ __forceinline__ __bf16 f2bf(float f) {
  unsigned short h = f2bf_u(f);
  return __builtin_bit_cast(__bf16, h);
}

__device__ __forceinline__ f32x8 wmma_bf16(bf16x16 a, bf16x16 b, f32x8 c) {
  // (neg_a, A, neg_b, B, c_mod, C, reuse_a, reuse_b)
  return __builtin_amdgcn_wmma_f32_16x16x32_bf16(false, a, false, b, (short)0, c,
                                                 false, false);
}

#if USE_ASYNC_LDS
// Builtin prototype (from hipcc diagnostic): pointers to GCC vector
// 'int __attribute__((vector_size(16)))' -- AS1 (global) src, AS3 (LDS) dst.
typedef int v4i_gcc __attribute__((vector_size(16)));
typedef __attribute__((address_space(1))) v4i_gcc as1_v4i;
typedef __attribute__((address_space(3))) v4i_gcc as3_v4i;

__device__ __forceinline__ void g2l_async16(const __bf16* g, __bf16* l) {
  // 16 bytes per lane, global -> LDS, no VGPR round trip (ASYNCcnt tracked)
  __builtin_amdgcn_global_load_async_to_lds_b128((as1_v4i*)g, (as3_v4i*)l, 0, 0);
}
__device__ __forceinline__ void wait_async0() {
#if __has_builtin(__builtin_amdgcn_s_wait_asynccnt)
  __builtin_amdgcn_s_wait_asynccnt(0);
#else
  asm volatile("s_wait_asynccnt 0x0" ::: "memory");
#endif
}
#endif

// ---------------------------------------------------------------------------
// fp32 -> bf16 cast (4 elements / thread)
// ---------------------------------------------------------------------------
__global__ __launch_bounds__(256)
void cast_f32_to_bf16(const float* __restrict__ src, __bf16* __restrict__ dst, int n4) {
  int i = blockIdx.x * blockDim.x + threadIdx.x;
  if (i >= n4) return;
  float4 f = ((const float4*)src)[i];
  unsigned long long p =  (unsigned long long)f2bf_u(f.x)
                       | ((unsigned long long)f2bf_u(f.y) << 16)
                       | ((unsigned long long)f2bf_u(f.z) << 32)
                       | ((unsigned long long)f2bf_u(f.w) << 48);
  ((unsigned long long*)dst)[i] = p;
}

// ---------------------------------------------------------------------------
// One 32-wide K slab of WMMAs from staged LDS tiles (8 WMMAs per wave).
// ---------------------------------------------------------------------------
__device__ __forceinline__ void mma_slab(const __bf16* sAbuf, const __bf16* sWbuf,
                                         f32x8 acc[4][2],
                                         int wm, int wn, int r, int hl)
{
  // A fragments (16x32 bf16): lane<16 holds K 0-7 & 16-23 of row r,
  // lane>=16 holds K 8-15 & 24-31 (ISA 7.12.2).
  Frag16 af[4];
  for (int mi = 0; mi < 4; ++mi) {
    const __bf16* p = sAbuf + (wm * 64 + mi * 16 + r) * 40;
    af[mi].q[0] = *(const uint4*)(p + hl * 8);
    af[mi].q[1] = *(const uint4*)(p + 16 + hl * 8);
  }
  // B fragments (32x16): lane = column n (= W row n), lanes 0-15 K 0-7 & 16-23,
  // lanes 16-31 K 8-15 & 24-31 (same K order as A).
  Frag16 bfg[2];
  for (int ni = 0; ni < 2; ++ni) {
    const __bf16* p = sWbuf + (wn * 32 + ni * 16 + r) * 40 + hl * 8;
    bfg[ni].q[0] = *(const uint4*)p;
    bfg[ni].q[1] = *(const uint4*)(p + 16);
  }
  for (int mi = 0; mi < 4; ++mi)
    for (int ni = 0; ni < 2; ++ni)
      acc[mi][ni] = wmma_bf16(af[mi].v, bfg[ni].v, acc[mi][ni]);
}

// ---------------------------------------------------------------------------
// C[M,N] = A[M,K] @ W[N,K]^T   (torch Linear semantics)
// Block: 256 thr = 8 waves; block tile 128x128; wave tile 64x32; K-step 32.
// mode 0: store bf16 row-major (M,N)         (Q, K projections)
// mode 1: store bf16 transposed per head:    Vt[(b*16+h)*64+d][s]  (V proj)
// mode 2: store f32 + bias row-major         (output projection)
// ---------------------------------------------------------------------------
__global__ __launch_bounds__(256)
void gemm_xwt(const __bf16* __restrict__ A, const __bf16* __restrict__ W,
              void* __restrict__ Cout, const float* __restrict__ bias,
              int M, int N, int K, int mode)
{
#if USE_ASYNC_LDS
  __shared__ __align__(16) __bf16 sA[2][128 * 40];   // double-buffered
  __shared__ __align__(16) __bf16 sW[2][128 * 40];
#else
  __shared__ __align__(16) __bf16 sA[1][128 * 40];
  __shared__ __align__(16) __bf16 sW[1][128 * 40];
#endif

  const int tid  = threadIdx.x;
  const int lane = tid & 31;
  const int wave = tid >> 5;
  const int wm   = wave >> 2;        // 0..1  (M)
  const int wn   = wave & 3;         // 0..3  (N)
  const int m0   = blockIdx.y * 128;
  const int n0   = blockIdx.x * 128;
  const int r    = lane & 15;
  const int hl   = lane >> 4;

  f32x8 acc[4][2];
  for (int mi = 0; mi < 4; ++mi)
    for (int ni = 0; ni < 2; ++ni)
      for (int e = 0; e < 8; ++e) acc[mi][ni][e] = 0.0f;

  const int ldr = tid >> 1;            // 0..127 (tile row)
  const int seg = (tid & 1) << 4;      // 0 or 16 (element offset in K slab)
  const __bf16* gA = A + (size_t)(m0 + ldr) * K + seg;
  const __bf16* gW = W + (size_t)(n0 + ldr) * K + seg;

#if USE_ASYNC_LDS
  constexpr int BUF = 128 * 40;
  __bf16* lA = &sA[0][ldr * 40 + seg];
  __bf16* lW = &sW[0][ldr * 40 + seg];

  // prologue: async-copy slab 0
  g2l_async16(gA,     lA);
  g2l_async16(gA + 8, lA + 8);
  g2l_async16(gW,     lW);
  g2l_async16(gW + 8, lW + 8);
  wait_async0();
  __syncthreads();

  int cur = 0;
  for (int k0 = 0; k0 < K; k0 += 32) {
    const int nxt = cur ^ 1;
    if (k0 + 32 < K) {                 // async-prefetch next slab into other buf
      g2l_async16(gA + k0 + 32, lA + nxt * BUF);
      g2l_async16(gA + k0 + 40, lA + nxt * BUF + 8);
      g2l_async16(gW + k0 + 32, lW + nxt * BUF);
      g2l_async16(gW + k0 + 40, lW + nxt * BUF + 8);
    }
    mma_slab(&sA[cur][0], &sW[cur][0], acc, wm, wn, r, hl);  // overlaps copy
    wait_async0();                     // own async writes to LDS landed
    __syncthreads();                   // everyone done reading cur / writing nxt
    cur = nxt;
  }
#else
  for (int k0 = 0; k0 < K; k0 += 32) {
    uint4 a0 = *(const uint4*)(gA + k0);
    uint4 a1 = *(const uint4*)(gA + k0 + 8);
    uint4 w0 = *(const uint4*)(gW + k0);
    uint4 w1 = *(const uint4*)(gW + k0 + 8);
    if (k0 + 32 < K) {
      __builtin_prefetch(gA + k0 + 32, 0, 1);
      __builtin_prefetch(gW + k0 + 32, 0, 1);
    }
    __syncthreads();                   // previous slab fully consumed
    *(uint4*)(&sA[0][ldr * 40 + seg    ]) = a0;
    *(uint4*)(&sA[0][ldr * 40 + seg + 8]) = a1;
    *(uint4*)(&sW[0][ldr * 40 + seg    ]) = w0;
    *(uint4*)(&sW[0][ldr * 40 + seg + 8]) = w1;
    __syncthreads();
    mma_slab(&sA[0][0], &sW[0][0], acc, wm, wn, r, hl);
  }
#endif

  // Epilogue: C layout -> VGPR e: row = base + e + 8*hl, col = lane&15.
  for (int mi = 0; mi < 4; ++mi) {
    const int mbase = m0 + wm * 64 + mi * 16 + hl * 8;
    for (int ni = 0; ni < 2; ++ni) {
      const int n = n0 + wn * 32 + ni * 16 + r;
      for (int e = 0; e < 8; ++e) {
        const int m = mbase + e;
        float v = acc[mi][ni][e];
        if (mode == 0) {
          ((__bf16*)Cout)[(size_t)m * N + n] = f2bf(v);
        } else if (mode == 1) {
          const int b = m >> 11, s = m & 2047;   // S = 2048
          const int h = n >> 6,  d = n & 63;     // dk = 64
          ((__bf16*)Cout)[(((size_t)(b * 16 + h) * 64 + d) << 11) + s] = f2bf(v);
        } else {
          ((float*)Cout)[(size_t)m * N + n] = v + bias[n];
        }
      }
    }
  }
}

// ---------------------------------------------------------------------------
// Flash attention: one wave per 16 queries of one (b,h); 32 keys / iteration.
// scores = Q(16x64) @ Kt  (2 WMMAs per 16-key tile); online softmax;
// P(16x32) @ V -> 4 WMMAs into 16x64 f32 accumulator.
// Q, K read directly (bf16, contiguous per lane); V read from Vt (dk x S).
// P is transposed C-layout -> A-layout through a private per-wave LDS tile.
// ---------------------------------------------------------------------------
__global__ __launch_bounds__(256)
void flash_attn(const __bf16* __restrict__ Q, const __bf16* __restrict__ K,
                const __bf16* __restrict__ Vt, const int* __restrict__ mask,
                __bf16* __restrict__ ctx)
{
  const int S = 2048, H = 1024, DK = 64, NH = 16;
  __shared__ __align__(16) __bf16 sP[8 * 16 * 40];

  const int lane = threadIdx.x & 31;
  const int wave = threadIdx.x >> 5;
  const int bid  = blockIdx.x;
  const int qt   = bid & 15;            // 16 q-tiles of 128 per (b,h)
  const int bh   = bid >> 4;            // b*NH + h
  const int h    = bh & (NH - 1);
  const int b    = bh >> 4;
  const int q0   = qt * 128 + wave * 16;
  const int r    = lane & 15;
  const int hl   = lane >> 4;

  __bf16* myP = &sP[wave * 16 * 40];

  // Q fragments for dk chunks 0..31 and 32..63 (A-matrix layout)
  Frag16 qf[2];
  const __bf16* qbase = Q + (size_t)(b * S + q0 + r) * H + h * DK;
  for (int c = 0; c < 2; ++c) {
    qf[c].q[0] = *(const uint4*)(qbase + c * 32 + hl * 8);
    qf[c].q[1] = *(const uint4*)(qbase + c * 32 + 16 + hl * 8);
  }

  f32x8 o[4];
  for (int nc = 0; nc < 4; ++nc)
    for (int e = 0; e < 8; ++e) o[nc][e] = 0.0f;
  float mrun[8], lrun[8];
  for (int e = 0; e < 8; ++e) { mrun[e] = -1e30f; lrun[e] = 0.0f; }

  const int* mrow = mask + b * S;
  const float scale = 0.125f;           // 1/sqrt(64)

  for (int k0 = 0; k0 < S; k0 += 32) {
    // two 16-key score tiles
    f32x8 sc[2];
    for (int t = 0; t < 2; ++t) {
      f32x8 a;
      for (int e = 0; e < 8; ++e) a[e] = 0.0f;
      for (int c = 0; c < 2; ++c) {
        Frag16 kf;  // B-matrix: lane = key column, K-dim = dk chunk
        const __bf16* kp = K + (size_t)(b * S + k0 + t * 16 + r) * H
                             + h * DK + c * 32 + hl * 8;
        kf.q[0] = *(const uint4*)kp;
        kf.q[1] = *(const uint4*)(kp + 16);
        a = wmma_bf16(qf[c].v, kf.v, a);
      }
      sc[t] = a;
    }

    const int mv0 = mrow[k0 + r];        // per key column (lane) mask
    const int mv1 = mrow[k0 + 16 + r];

    float alpha[8];
    for (int i = 0; i < 8; ++i) {
      float s0 = sc[0][i] * scale; if (mv0 == 0) s0 = -1e9f;
      float s1 = sc[1][i] * scale; if (mv1 == 0) s1 = -1e9f;
      float rm = fmaxf(s0, s1);
      for (int d = 1; d < 16; d <<= 1) rm = fmaxf(rm, __shfl_xor(rm, d, 32));
      const float mnew = fmaxf(mrun[i], rm);
      const float al = __expf(mrun[i] - mnew);
      const float p0 = __expf(s0 - mnew);
      const float p1 = __expf(s1 - mnew);
      float rs = p0 + p1;
      for (int d = 1; d < 16; d <<= 1) rs += __shfl_xor(rs, d, 32);
      lrun[i] = lrun[i] * al + rs;
      mrun[i] = mnew;
      alpha[i] = al;
      // P element (row = i + 8*hl, key col = r / 16+r) -> LDS
      myP[(i + 8 * hl) * 40 + r]      = f2bf(p0);
      myP[(i + 8 * hl) * 40 + 16 + r] = f2bf(p1);
    }

    // rescale running output
    for (int nc = 0; nc < 4; ++nc)
      for (int i = 0; i < 8; ++i) o[nc][i] *= alpha[i];

    // make the wave's LDS P stores visible before fragment reloads
    asm volatile("s_wait_dscnt 0" ::: "memory");

    // P as A-matrix fragment (16 queries x 32 keys)
    Frag16 pf;
    {
      const __bf16* p = myP + r * 40;
      pf.q[0] = *(const uint4*)(p + hl * 8);
      pf.q[1] = *(const uint4*)(p + 16 + hl * 8);
    }

    // P @ V over 4 dk chunks; V columns contiguous thanks to Vt layout
    for (int nc = 0; nc < 4; ++nc) {
      Frag16 vf;
      const __bf16* vp = Vt + (size_t)(bh * DK + nc * 16 + r) * S + k0 + hl * 8;
      vf.q[0] = *(const uint4*)vp;
      vf.q[1] = *(const uint4*)(vp + 16);
      o[nc] = wmma_bf16(pf.v, vf.v, o[nc]);
    }
  }

  // finalize: divide by softmax denominator, store bf16 ctx (B*S, H)
  for (int nc = 0; nc < 4; ++nc)
    for (int i = 0; i < 8; ++i) {
      const float v = o[nc][i] / lrun[i];
      const int row = b * S + q0 + i + 8 * hl;
      const int col = h * DK + nc * 16 + r;
      ctx[(size_t)row * H + col] = f2bf(v);
    }
}

// ---------------------------------------------------------------------------
// Host launcher.  Inputs: x, mask, Wq, Wk, Wv, Wo, bo.  Output: (B,S,H) f32.
// Workspace (~88 MB): bf16 x, bf16 weights, bf16 Q/K, bf16 V^T, bf16 ctx.
// ---------------------------------------------------------------------------
extern "C" void kernel_launch(void* const* d_in, const int* in_sizes, int n_in,
                              void* d_out, int out_size, void* d_ws, size_t ws_size,
                              hipStream_t stream) {
  (void)in_sizes; (void)n_in; (void)out_size; (void)ws_size;
  const int B = 4, S = 2048, H = 1024;
  const int M = B * S;                  // 8192

  const float* x    = (const float*)d_in[0];
  const int*   mask = (const int*)  d_in[1];
  const float* Wq   = (const float*)d_in[2];
  const float* Wk   = (const float*)d_in[3];
  const float* Wv   = (const float*)d_in[4];
  const float* Wo   = (const float*)d_in[5];
  const float* bo   = (const float*)d_in[6];
  float* out = (float*)d_out;

  char* ws = (char*)d_ws;
  const size_t xhN = (size_t)M * H;     // 8,388,608
  const size_t wN  = (size_t)H * H;     // 1,048,576
  __bf16* xh  = (__bf16*)(ws);
  __bf16* Wqh = (__bf16*)(ws + 16777216);
  __bf16* Wkh = (__bf16*)(ws + 18874368);
  __bf16* Wvh = (__bf16*)(ws + 20971520);
  __bf16* Woh = (__bf16*)(ws + 23068672);
  __bf16* Qh  = (__bf16*)(ws + 25165824);
  __bf16* Kh  = (__bf16*)(ws + 41943040);
  __bf16* Vt  = (__bf16*)(ws + 58720256);
  __bf16* Ctx = (__bf16*)(ws + 75497472);

  // casts
  cast_f32_to_bf16<<<(int)(xhN / 4 / 256), 256, 0, stream>>>(x,  xh,  (int)(xhN / 4));
  cast_f32_to_bf16<<<(int)(wN  / 4 / 256), 256, 0, stream>>>(Wq, Wqh, (int)(wN / 4));
  cast_f32_to_bf16<<<(int)(wN  / 4 / 256), 256, 0, stream>>>(Wk, Wkh, (int)(wN / 4));
  cast_f32_to_bf16<<<(int)(wN  / 4 / 256), 256, 0, stream>>>(Wv, Wvh, (int)(wN / 4));
  cast_f32_to_bf16<<<(int)(wN  / 4 / 256), 256, 0, stream>>>(Wo, Woh, (int)(wN / 4));

  // projections: Q, K row-major bf16; V stored transposed per head
  dim3 ggrid(H / 128, M / 128);         // (8, 64)
  gemm_xwt<<<ggrid, 256, 0, stream>>>(xh, Wqh, (void*)Qh, nullptr, M, H, H, 0);
  gemm_xwt<<<ggrid, 256, 0, stream>>>(xh, Wkh, (void*)Kh, nullptr, M, H, H, 0);
  gemm_xwt<<<ggrid, 256, 0, stream>>>(xh, Wvh, (void*)Vt, nullptr, M, H, H, 1);

  // flash attention: B*nh*(S/128) blocks
  flash_attn<<<4 * 16 * (S / 128), 256, 0, stream>>>(Qh, Kh, Vt, mask, Ctx);

  // output projection with bias, f32 out
  gemm_xwt<<<ggrid, 256, 0, stream>>>(Ctx, Woh, (void*)out, bo, M, H, H, 2);
}
